// Attention_43963285242565
// MI455X (gfx1250) — matrix operand order repaired and checked by reading, not compiled, not run on hardware
//
#include <hip/hip_runtime.h>
#include <hip/hip_bf16.h>

// ---------------------------------------------------------------------------
// Problem constants (B=1)
// ---------------------------------------------------------------------------
#define LQ    2048     // sequence length
#define DM    2048     // model dim
#define NHq   32       // query heads
#define NKVh  8        // kv heads
#define HDh   64       // head dim
#define NQKV  3072     // NH*HD + 2*NKV*HD
#define SCALE_Q 0.125f // HD^-0.5

typedef __bf16 bf16x16 __attribute__((ext_vector_type(16)));
typedef float  f32x8   __attribute__((ext_vector_type(8)));

union Frag16 { bf16x16 v; unsigned int u[8]; };

struct __align__(16) B128 { unsigned int w0, w1, w2, w3; };

__device__ __forceinline__ unsigned short f2bf(float f) {
  unsigned int u = __float_as_uint(f);
  u += 0x7FFFu + ((u >> 16) & 1u);      // round-to-nearest-even
  return (unsigned short)(u >> 16);
}
__device__ __forceinline__ unsigned int pack2bf(float a, float b) {
  return (unsigned int)f2bf(a) | ((unsigned int)f2bf(b) << 16);
}

// low 32 bits of a generic pointer into LDS == LDS byte offset
__device__ __forceinline__ unsigned lds_off(const void* p) {
  return (unsigned)(unsigned long long)(uintptr_t)p;
}
// per-lane async DMA: 16 bytes global -> LDS, tracked by ASYNCcnt (GVS mode)
__device__ __forceinline__ void async_b128(unsigned loff, unsigned goff, const void* gbase) {
  asm volatile("global_load_async_to_lds_b128 %0, %1, %2"
               :: "v"(loff), "v"(goff), "s"((unsigned long long)(uintptr_t)gbase)
               : "memory");
}
__device__ __forceinline__ void wait_async0() {
  asm volatile("s_wait_asynccnt 0" ::: "memory");
}
// 16-byte aligned LDS fragment chunk load (lowers to ds_load_b128)
__device__ __forceinline__ void ld4(const unsigned short* p, unsigned int* dst) {
  B128 t = *(const B128*)p;
  dst[0] = t.w0; dst[1] = t.w1; dst[2] = t.w2; dst[3] = t.w3;
}

// ---------------------------------------------------------------------------
// Kernel 0: f32 -> packed bf16 cast
// ---------------------------------------------------------------------------
__global__ void cast_f32_bf16(const float* __restrict__ in,
                              unsigned int* __restrict__ out, int npairs) {
  int stride = gridDim.x * blockDim.x;
  for (int i = blockIdx.x * blockDim.x + threadIdx.x; i < npairs; i += stride)
    out[i] = pack2bf(in[2 * i], in[2 * i + 1]);
}

// ---------------------------------------------------------------------------
// Kernel 1/4: C(MxN, f32) = A(MxK, bf16) * Bw(NxK, bf16)^T  via v_wmma bf16
// 256 threads = 8 waves (2x4), 128x128 tile, K-step 32, double-buffered
// async global->LDS tile staging.
// ---------------------------------------------------------------------------
#define BM 128
#define BN 128
#define BKs 32
#define LDT 40   // LDS row stride (bf16): 80 B = 16-byte multiple, bank-clean

__device__ __forceinline__ void stage_tile_async(const unsigned short* g, int rowbase,
                                                 int K, int k0,
                                                 unsigned short* lds, int tid) {
#pragma unroll
  for (int i = 0; i < 2; ++i) {
    int idx = tid + i * 256;            // 512 x 16-byte chunks per 128x32 tile
    int row = idx >> 2;
    int c8  = idx & 3;
    unsigned goff = (unsigned)(((rowbase + row) * K + k0 + c8 * 8) * 2);
    async_b128(lds_off(&lds[row * LDT + c8 * 8]), goff, g);
  }
}

__launch_bounds__(256)
__global__ void gemm_bf16_wmma(const unsigned short* __restrict__ A,
                               const unsigned short* __restrict__ Bw,
                               float* __restrict__ C,
                               int M, int N, int K) {
  __shared__ __align__(16) unsigned short ldsA[2][BM * LDT];
  __shared__ __align__(16) unsigned short ldsB[2][BN * LDT];
  const int tid  = threadIdx.x;
  const int lane = tid & 31;
  const int wave = tid >> 5;
  const int ln = lane & 15;
  const int hi = lane >> 4;
  const int wm = wave >> 2;          // 0..1 -> 64 rows each
  const int wn = wave & 3;           // 0..3 -> 32 cols each
  const int m0 = blockIdx.y * BM;
  const int n0 = blockIdx.x * BN;

  f32x8 acc[4][2];
#pragma unroll
  for (int mf = 0; mf < 4; ++mf)
#pragma unroll
    for (int nf = 0; nf < 2; ++nf)
#pragma unroll
      for (int e = 0; e < 8; ++e) acc[mf][nf][e] = 0.0f;

  const int nk = K / BKs;
  stage_tile_async(A,  m0, K, 0, ldsA[0], tid);
  stage_tile_async(Bw, n0, K, 0, ldsB[0], tid);

#pragma unroll 1
  for (int j = 0; j < nk; ++j) {
    wait_async0();                      // my buf[j&1] DMAs done
    __syncthreads();                    // everyone's done -> LDS coherent
    if (j + 1 < nk) {                   // prefetch next tile during compute
      stage_tile_async(A,  m0, K, (j + 1) * BKs, ldsA[(j + 1) & 1], tid);
      stage_tile_async(Bw, n0, K, (j + 1) * BKs, ldsB[(j + 1) & 1], tid);
    }
    const unsigned short* la = ldsA[j & 1];
    const unsigned short* lb = ldsB[j & 1];

    Frag16 af[4], bfm[2];
#pragma unroll
    for (int mf = 0; mf < 4; ++mf) {    // A layout: M=ln, K = 8*hi+(i&7)+16*(i>=8)
      const unsigned short* base = &la[(wm * 64 + mf * 16 + ln) * LDT + 8 * hi];
      ld4(base,      &af[mf].u[0]);
      ld4(base + 16, &af[mf].u[4]);
    }
#pragma unroll
    for (int nf = 0; nf < 2; ++nf) {    // B layout: N=ln, K = 8*hi+(i&7)+16*(i>=8)
      const unsigned short* base = &lb[(wn * 32 + nf * 16 + ln) * LDT + 8 * hi];
      ld4(base,      &bfm[nf].u[0]);
      ld4(base + 16, &bfm[nf].u[4]);
    }
#pragma unroll
    for (int mf = 0; mf < 4; ++mf)
#pragma unroll
      for (int nf = 0; nf < 2; ++nf)
        acc[mf][nf] = __builtin_amdgcn_wmma_f32_16x16x32_bf16(
            false, af[mf].v, false, bfm[nf].v, (short)0, acc[mf][nf], false, false);
    __syncthreads();                    // reads done before buffer reuse
  }

#pragma unroll
  for (int mf = 0; mf < 4; ++mf)
#pragma unroll
    for (int nf = 0; nf < 2; ++nf)
#pragma unroll
      for (int r = 0; r < 8; ++r) {     // C layout: M = r + 8*hi, N = ln
        int row = m0 + wm * 64 + mf * 16 + r + 8 * hi;
        int col = n0 + wn * 32 + nf * 16 + ln;
        C[(unsigned)(row * N + col)] = acc[mf][nf][r];
      }
}

// ---------------------------------------------------------------------------
// Kernel 2: RoPE + RMSNorm + bf16 cast + head-major repack.
// One wave32 per 64-wide row; lane p owns the (2p, 2p+1) pair.
// ---------------------------------------------------------------------------
__launch_bounds__(256)
__global__ void rope_rms_cast(const float* __restrict__ qkv,
                              const float* __restrict__ qg,
                              const float* __restrict__ kg,
                              unsigned int* __restrict__ qb,
                              unsigned int* __restrict__ kb,
                              unsigned int* __restrict__ vb) {
  const int gwave = (blockIdx.x * blockDim.x + threadIdx.x) >> 5;
  const int p = threadIdx.x & 31;
  const int slot = gwave % 48;        // 0..31 q, 32..39 k, 40..47 v
  const int l = gwave / 48;
  if (l >= LQ) return;

  if (slot < NHq + NKVh) {            // q or k rows: rope + rmsnorm
    const bool isq = slot < NHq;
    const int head = isq ? slot : slot - NHq;
    const int base = isq ? head * HDh : NHq * HDh + head * HDh;
    const float* r = qkv + (unsigned)(l * NQKV + base);
    float x1 = r[2 * p], x2 = r[2 * p + 1];
    float inv = __expf(-logf(50000.0f) * (float)p * (1.0f / 32.0f)); // theta^(-p/32)
    float s, c;
    __sincosf((float)l * inv, &s, &c);
    float r1 = x1 * c - x2 * s;
    float r2 = x1 * s + x2 * c;
    float ss = r1 * r1 + r2 * r2;
#pragma unroll
    for (int msk = 16; msk >= 1; msk >>= 1) ss += __shfl_xor(ss, msk, 32);
    float rms = rsqrtf(ss * (1.0f / 64.0f) + 1e-5f);
    const float* g = isq ? qg : kg;
    float sc = isq ? SCALE_Q : 1.0f;  // fold attention scale into q
    float y1 = r1 * rms * g[2 * p] * sc;
    float y2 = r2 * rms * g[2 * p + 1] * sc;
    unsigned int* dst = isq ? qb : kb;
    dst[(unsigned)((head * LQ + l) * 32 + p)] = pack2bf(y1, y2);
  } else {                            // v rows: straight cast
    const int kv = slot - NHq - NKVh;
    const float* r = qkv + (unsigned)(l * NQKV + (NHq + NKVh) * HDh + kv * HDh);
    vb[(unsigned)((kv * LQ + l) * 32 + p)] = pack2bf(r[2 * p], r[2 * p + 1]);
  }
}

// ---------------------------------------------------------------------------
// Kernel 3: flash attention, causal, GQA (4 q-heads per kv-head).
// Block = 128 threads (4 waves) per (head, 64-row q tile); each wave owns 16
// q rows. KV in 64-wide chunks, double-buffered (K async DMA, V transposed
// manually), online softmax.
// ---------------------------------------------------------------------------
#define FST 72   // LDS row stride (bf16): 144 B = 16-byte multiple

__device__ __forceinline__ void stage_k_async(const unsigned short* kb, int kvh, int k0,
                                              unsigned short* ldsK, int tid) {
#pragma unroll
  for (int i = 0; i < 4; ++i) {
    int idx = tid + i * 128;            // 512 x 16-byte chunks per 64x64 tile
    int row = idx >> 3, c8 = idx & 7;
    unsigned goff = (unsigned)(((kvh * LQ + k0 + row) * HDh + c8 * 8) * 2);
    async_b128(lds_off(&ldsK[row * FST + c8 * 8]), goff, kb);
  }
}
__device__ __forceinline__ void stage_v_manual(const unsigned short* vb, int kvh, int k0,
                                               unsigned short* ldsV, int tid) {
#pragma unroll
  for (int i = 0; i < 16; ++i) {
    int idx = tid + i * 128;
    int row = idx >> 5, c2 = idx & 31;
    unsigned int vv = ((const unsigned int*)(vb + (unsigned)((kvh * LQ + k0 + row) * HDh)))[c2];
    ldsV[(c2 * 2 + 0) * FST + row] = (unsigned short)(vv & 0xFFFFu);
    ldsV[(c2 * 2 + 1) * FST + row] = (unsigned short)(vv >> 16);
  }
}

__launch_bounds__(128)
__global__ void flash_attn(const unsigned short* __restrict__ qb,
                           const unsigned short* __restrict__ kb,
                           const unsigned short* __restrict__ vb,
                           unsigned short* __restrict__ ctx) {
  __shared__ __align__(16) unsigned short ldsQ[64 * FST];
  __shared__ __align__(16) unsigned short ldsK[2][64 * FST];
  __shared__ __align__(16) unsigned short ldsV[2][64 * FST]; // transposed [d][kv]
  __shared__ __align__(16) unsigned short ldsP[4][16 * FST]; // per-wave P staging
  const int h   = blockIdx.x;
  const int q0  = blockIdx.y * 64;
  const int kvh = h >> 2;
  const int tid = threadIdx.x;
  const int wave = tid >> 5;
  const int lane = tid & 31;
  const int ln = lane & 15;
  const int hi = lane >> 4;

  // stage Q (async) + first KV chunk
#pragma unroll
  for (int i = 0; i < 4; ++i) {
    int idx = tid + i * 128;
    int row = idx >> 3, c8 = idx & 7;
    unsigned goff = (unsigned)(((h * LQ + q0 + row) * HDh + c8 * 8) * 2);
    async_b128(lds_off(&ldsQ[row * FST + c8 * 8]), goff, qb);
  }
  stage_k_async(kb, kvh, 0, ldsK[0], tid);
  stage_v_manual(vb, kvh, 0, ldsV[0], tid);
  wait_async0();
  __syncthreads();

  Frag16 qa[2];                        // Q A-frags, K halves 0-31 / 32-63
#pragma unroll
  for (int kk = 0; kk < 2; ++kk) {
    const unsigned short* base = &ldsQ[(wave * 16 + ln) * FST + kk * 32 + 8 * hi];
    ld4(base,      &qa[kk].u[0]);
    ld4(base + 16, &qa[kk].u[4]);
  }

  f32x8 o[4];
  float mrow[8], lrow[8];
#pragma unroll
  for (int nf = 0; nf < 4; ++nf)
#pragma unroll
    for (int e = 0; e < 8; ++e) o[nf][e] = 0.0f;
#pragma unroll
  for (int r = 0; r < 8; ++r) { mrow[r] = -1e30f; lrow[r] = 0.0f; }

  const int nchunks = (q0 >> 6) + 1;   // causal: only chunks up to diagonal
#pragma unroll 1
  for (int j = 0; j < nchunks; ++j) {
    const int k0 = j * 64;
    if (j > 0) {
      wait_async0();                   // buf[j&1] K DMA done
      __syncthreads();                 // + everyone's V stores visible
    }
    if (j + 1 < nchunks) {             // prefetch next chunk during compute
      stage_k_async(kb, kvh, k0 + 64, ldsK[(j + 1) & 1], tid);
      stage_v_manual(vb, kvh, k0 + 64, ldsV[(j + 1) & 1], tid);
    }
    const unsigned short* lK = ldsK[j & 1];
    const unsigned short* lV = ldsV[j & 1];

    // S = Q * K^T  (16 x 64 per wave, 4 column fragments)
    f32x8 sfr[4];
#pragma unroll
    for (int nf = 0; nf < 4; ++nf) {
      Frag16 b0, b1;
      const unsigned short* base = &lK[(nf * 16 + ln) * FST + 8 * hi];
      ld4(base,          &b0.u[0]);
      ld4(base + 16,     &b0.u[4]);
      ld4(base + 32,     &b1.u[0]);
      ld4(base + 48,     &b1.u[4]);
      f32x8 s;
#pragma unroll
      for (int e = 0; e < 8; ++e) s[e] = 0.0f;
      s = __builtin_amdgcn_wmma_f32_16x16x32_bf16(false, qa[0].v, false, b0.v, (short)0, s, false, false);
      s = __builtin_amdgcn_wmma_f32_16x16x32_bf16(false, qa[1].v, false, b1.v, (short)0, s, false, false);
      sfr[nf] = s;
    }

    if (k0 == q0) {                    // diagonal chunk: elementwise causal mask
#pragma unroll
      for (int nf = 0; nf < 4; ++nf)
#pragma unroll
        for (int r = 0; r < 8; ++r) {
          int kcol = nf * 16 + ln;
          int qrow = wave * 16 + r + 8 * hi;
          if (kcol > qrow) sfr[nf][r] = -1e30f;
        }
    }

    // online softmax stats (rows live per 16-lane half -> 4-step xor reduce)
    float nm[8], alpha[8];
#pragma unroll
    for (int r = 0; r < 8; ++r) {
      float v = fmaxf(fmaxf(sfr[0][r], sfr[1][r]), fmaxf(sfr[2][r], sfr[3][r]));
#pragma unroll
      for (int msk = 8; msk >= 1; msk >>= 1) v = fmaxf(v, __shfl_xor(v, msk, 32));
      nm[r] = fmaxf(mrow[r], v);
      alpha[r] = __expf(mrow[r] - nm[r]);
      mrow[r] = nm[r];
    }
    float rs[8];
#pragma unroll
    for (int r = 0; r < 8; ++r) rs[r] = 0.0f;
#pragma unroll
    for (int nf = 0; nf < 4; ++nf)
#pragma unroll
      for (int r = 0; r < 8; ++r) {
        float pv = __expf(sfr[nf][r] - nm[r]);
        sfr[nf][r] = pv;
        rs[r] += pv;
      }
#pragma unroll
    for (int r = 0; r < 8; ++r) {
      float v = rs[r];
#pragma unroll
      for (int msk = 8; msk >= 1; msk >>= 1) v += __shfl_xor(v, msk, 32);
      lrow[r] = lrow[r] * alpha[r] + v;
    }
#pragma unroll
    for (int nf = 0; nf < 4; ++nf)
#pragma unroll
      for (int r = 0; r < 8; ++r) o[nf][r] *= alpha[r];

    // P: C-layout -> A-layout via per-wave LDS round-trip (bf16)
#pragma unroll
    for (int nf = 0; nf < 4; ++nf)
#pragma unroll
      for (int r = 0; r < 8; ++r)
        ldsP[wave][(r + 8 * hi) * FST + nf * 16 + ln] = f2bf(sfr[nf][r]);

    Frag16 pa[2];
#pragma unroll
    for (int kk = 0; kk < 2; ++kk) {
      const unsigned short* base = &ldsP[wave][ln * FST + kk * 32 + 8 * hi];
      ld4(base,      &pa[kk].u[0]);
      ld4(base + 16, &pa[kk].u[4]);
    }

    // O += P * V   (V gathered from transposed LDS: B-layout, K = kv index)
#pragma unroll
    for (int nf = 0; nf < 4; ++nf) {
      Frag16 b0, b1;
      const unsigned short* base = &lV[(nf * 16 + ln) * FST + 8 * hi];
      ld4(base,      &b0.u[0]);
      ld4(base + 16, &b0.u[4]);
      ld4(base + 32, &b1.u[0]);
      ld4(base + 48, &b1.u[4]);
      o[nf] = __builtin_amdgcn_wmma_f32_16x16x32_bf16(false, pa[0].v, false, b0.v, (short)0, o[nf], false, false);
      o[nf] = __builtin_amdgcn_wmma_f32_16x16x32_bf16(false, pa[1].v, false, b1.v, (short)0, o[nf], false, false);
    }
  }

  // epilogue: normalize and store ctx[l][h*64+d] as bf16 for the final GEMM
#pragma unroll
  for (int nf = 0; nf < 4; ++nf)
#pragma unroll
    for (int r = 0; r < 8; ++r) {
      int qrow = q0 + wave * 16 + r + 8 * hi;
      float val = o[nf][r] / lrow[r];
      ctx[(unsigned)(qrow * DM + h * HDh + nf * 16 + ln)] = f2bf(val);
    }
}

// ---------------------------------------------------------------------------
// Launcher
// ---------------------------------------------------------------------------
extern "C" void kernel_launch(void* const* d_in, const int* in_sizes, int n_in,
                              void* d_out, int out_size, void* d_ws, size_t ws_size,
                              hipStream_t stream) {
  (void)in_sizes; (void)n_in; (void)out_size; (void)ws_size;
  const float* x     = (const float*)d_in[0];
  const float* w_qkv = (const float*)d_in[1];
  const float* w_out = (const float*)d_in[2];
  const float* qg    = (const float*)d_in[3];
  const float* kg    = (const float*)d_in[4];
  // d_in[5] = causal mask, implicit in flash kernel

  // workspace layout (64 MB total); ctx aliases xb (xb dead after GEMM1)
  char* ws = (char*)d_ws;
  unsigned short* xb    = (unsigned short*)(ws + 0);          //  8 MB
  unsigned short* wqkvb = (unsigned short*)(ws + 8388608);    // 12 MB
  unsigned short* woutb = (unsigned short*)(ws + 20971520);   //  8 MB
  float*          qkv   = (float*)         (ws + 29360128);   // 24 MB
  unsigned short* qb    = (unsigned short*)(ws + 54525952);   //  8 MB
  unsigned short* kb    = (unsigned short*)(ws + 62914560);   //  2 MB
  unsigned short* vb    = (unsigned short*)(ws + 65011712);   //  2 MB
  unsigned short* ctx   = xb;

  // 0) casts
  {
    int np = LQ * DM / 2;
    cast_f32_bf16<<<(np + 255) / 256, 256, 0, stream>>>(x, (unsigned int*)xb, np);
    np = NQKV * DM / 2;
    cast_f32_bf16<<<(np + 255) / 256, 256, 0, stream>>>(w_qkv, (unsigned int*)wqkvb, np);
    np = DM * DM / 2;
    cast_f32_bf16<<<(np + 255) / 256, 256, 0, stream>>>(w_out, (unsigned int*)woutb, np);
  }
  // 1) qkv = x @ w_qkv^T  (f32 accumulate)
  gemm_bf16_wmma<<<dim3(NQKV / BN, LQ / BM), 256, 0, stream>>>(xb, wqkvb, qkv, LQ, NQKV, DM);
  // 2) RoPE + RMSNorm + repack
  rope_rms_cast<<<(LQ * 48) / 8, 256, 0, stream>>>(qkv, qg, kg,
                                                   (unsigned int*)qb,
                                                   (unsigned int*)kb,
                                                   (unsigned int*)vb);
  // 3) flash attention
  flash_attn<<<dim3(NHq, LQ / 64), 128, 0, stream>>>(qb, kb, vb, ctx);
  // 4) out = ctx @ w_out^T
  gemm_bf16_wmma<<<dim3(DM / BN, LQ / BM), 256, 0, stream>>>(ctx, woutb, (float*)d_out, LQ, DM, DM);
}
